// AttnBlock_24146306138438
// MI455X (gfx1250) — matrix operand order repaired and checked by reading, not compiled, not run on hardware
//
#include <hip/hip_runtime.h>
#include <hip/hip_bf16.h>

#define BB 4
#define CC 512
#define HWN 4096
#define GR 32
#define CPG (CC / GR)      // 16 channels per group
#define EPSV 1e-6f

#define SALD 520           // padded bf16 row stride for staged GEMM A tiles
#define SQLD 520           // padded bf16 row stride for staged Q (64 x 512)
#define SSLD 260           // padded f32 row stride for scores (64 x 256)

typedef __bf16 bf16;
typedef __attribute__((ext_vector_type(16))) __bf16 bf16x16;
typedef __attribute__((ext_vector_type(8)))  float  f32x8;

union ABFrag { bf16x16 v; uint4 q[2]; };

// A-fragment (16x32 bf16, M x K) from row-major src (ld in elements).
// Lanes 0-15: row M=lane, VGPR0-3 = K0..7, VGPR4-7 = K16..23.
// Lanes 16-31: row M=lane-16, K8..15 and K24..31.
__device__ inline bf16x16 load_a_rm(const bf16* src, int ld, int lane) {
    int r = lane & 15, h = lane >> 4;
    const bf16* p = src + (size_t)r * ld + h * 8;
    ABFrag f;
    f.q[0] = *(const uint4*)(p);
    f.q[1] = *(const uint4*)(p + 16);
    return f.v;
}

// Same A layout, but source is f32 (converted on the fly to bf16).
__device__ inline bf16x16 load_a_f32(const float* src, int ld, int lane) {
    int r = lane & 15, h = lane >> 4;
    const float* p = src + (size_t)r * ld + h * 8;
    float4 a0 = *(const float4*)(p + 0),  a1 = *(const float4*)(p + 4);
    float4 b0 = *(const float4*)(p + 16), b1 = *(const float4*)(p + 20);
    bf16x16 v;
    v[0]=(bf16)a0.x;  v[1]=(bf16)a0.y;  v[2]=(bf16)a0.z;  v[3]=(bf16)a0.w;
    v[4]=(bf16)a1.x;  v[5]=(bf16)a1.y;  v[6]=(bf16)a1.z;  v[7]=(bf16)a1.w;
    v[8]=(bf16)b0.x;  v[9]=(bf16)b0.y;  v[10]=(bf16)b0.z; v[11]=(bf16)b0.w;
    v[12]=(bf16)b1.x; v[13]=(bf16)b1.y; v[14]=(bf16)b1.z; v[15]=(bf16)b1.w;
    return v;
}

// B-fragment (32x16 bf16, K x N), source row-major with row = N, col = K.
__device__ inline bf16x16 load_b_nk(const bf16* src, int ld, int lane) {
    int n = lane & 15, h = lane >> 4;
    const bf16* p = src + (size_t)n * ld + h * 8;
    ABFrag f;
    f.q[0] = *(const uint4*)(p);
    f.q[1] = *(const uint4*)(p + 16);
    return f.v;
}

__device__ inline f32x8 wmma_bf16(bf16x16 a, bf16x16 b, f32x8 c) {
    return __builtin_amdgcn_wmma_f32_16x16x32_bf16(false, a, false, b,
                                                   (short)0, c, false, false);
}

// 16B async global->LDS copy (ASYNCcnt-tracked)
__device__ inline void async_cp16(void* lds_dst, const void* gsrc) {
    unsigned int la = (unsigned int)(uintptr_t)lds_dst;
    unsigned long long ga = (unsigned long long)(uintptr_t)gsrc;
    asm volatile("global_load_async_to_lds_b128 %0, %1, off"
                 :: "v"(la), "v"(ga) : "memory");
}
__device__ inline void async_wait0() {
    asm volatile("s_wait_asynccnt 0x0" ::: "memory");
}

// ---------------- GroupNorm ----------------
__global__ void gn_stats_kernel(const float* __restrict__ x, float* __restrict__ stats) {
    int bg = blockIdx.x;                       // b*GR + g ; group block is contiguous
    const size_t n = (size_t)CPG * HWN;        // 65536
    const float4* p = (const float4*)(x + (size_t)bg * n);
    float s = 0.f, s2 = 0.f;
    for (size_t i = threadIdx.x; i < n / 4; i += blockDim.x) {
        float4 v = p[i];
        s  += v.x + v.y + v.z + v.w;
        s2 += v.x * v.x + v.y * v.y + v.z * v.z + v.w * v.w;
    }
    __shared__ float rs[256], rs2[256];
    rs[threadIdx.x] = s; rs2[threadIdx.x] = s2;
    __syncthreads();
    for (int off = 128; off > 0; off >>= 1) {
        if ((int)threadIdx.x < off) {
            rs[threadIdx.x]  += rs[threadIdx.x + off];
            rs2[threadIdx.x] += rs2[threadIdx.x + off];
        }
        __syncthreads();
    }
    if (threadIdx.x == 0) {
        float mean = rs[0] / (float)n;
        float var  = rs2[0] / (float)n - mean * mean;
        stats[bg * 2 + 0] = mean;
        stats[bg * 2 + 1] = rsqrtf(var + EPSV);
    }
}

__global__ void gn_apply_kernel(const float* __restrict__ x,
                                const float* __restrict__ gamma,
                                const float* __restrict__ beta,
                                const float* __restrict__ stats,
                                bf16* __restrict__ xn) {
    int bc = blockIdx.x;                       // b*CC + c
    int b = bc / CC, c = bc % CC;
    int g = c / CPG;
    float mean = stats[(b * GR + g) * 2 + 0];
    float rstd = stats[(b * GR + g) * 2 + 1];
    float ga = gamma[c], be = beta[c];
    const float* src = x + (size_t)bc * HWN;
    bf16* dst = xn + (size_t)b * HWN * CC + c;
    for (int i = threadIdx.x; i < HWN; i += blockDim.x) {
        float vn = (src[i] - mean) * rstd * ga + be;
        dst[(size_t)i * CC] = (bf16)vn;
    }
}

// ---------------- Weight fp32 -> bf16 ----------------
__global__ void wcvt_kernel(const float* __restrict__ wq, const float* __restrict__ wk,
                            const float* __restrict__ wv, const float* __restrict__ wp,
                            bf16* __restrict__ wb) {
    int tid = blockIdx.x * blockDim.x + threadIdx.x;
    const int N = CC * CC;
    if (tid >= 4 * N) return;
    const float* src = (tid < N) ? wq : (tid < 2 * N) ? wk : (tid < 3 * N) ? wv : wp;
    wb[tid] = (bf16)src[tid & (N - 1)];
}

// ---------------- QKV projection GEMM (M = 128 rows / block) ----------------
__global__ void qkv_kernel(const bf16* __restrict__ xn, const bf16* __restrict__ wb,
                           const float* __restrict__ bq, const float* __restrict__ bk,
                           const float* __restrict__ bv,
                           bf16* __restrict__ qo, bf16* __restrict__ ko,
                           bf16* __restrict__ vo) {
    int blk = blockIdx.x;
    int b   = blk >> 5;            // HWN/128 = 32 row tiles per batch
    int hwb = (blk & 31) * 128;
    __shared__ __align__(16) bf16 sA[128 * SALD];
    {   // async-stage 128 contiguous rows of Xn (128KB), padded rows
        const bf16* g = xn + ((size_t)b * HWN + hwb) * CC;
        for (int i = threadIdx.x; i < 128 * (CC / 8); i += blockDim.x) {
            int row = i >> 6, cw = i & 63;
            async_cp16(sA + row * SALD + cw * 8, g + (size_t)row * CC + cw * 8);
        }
        async_wait0();
    }
    __syncthreads();
    int wave = threadIdx.x >> 5, lane = threadIdx.x & 31;
    const float scale = rsqrtf((float)CC);
    for (int tt = 0; tt < 12; ++tt) {          // 96 chains over 8 waves
        asm volatile("" ::: "memory");         // keep A-frag ds loads in-loop
        int t   = wave + tt * 8;
        int mtx = t >> 5;
        int cb  = (t & 31) * 16;
        const bf16* W = wb + (size_t)mtx * CC * CC + (size_t)cb * CC;
        f32x8 acc[8];
        for (int mt = 0; mt < 8; ++mt) acc[mt] = (f32x8){};
        for (int kb = 0; kb < CC; kb += 32) {
            bf16x16 bm = load_b_nk(W + kb, CC, lane);
            for (int mt = 0; mt < 8; ++mt) {
                bf16x16 a = load_a_rm(sA + mt * 16 * SALD + kb, SALD, lane);
                acc[mt] = wmma_bf16(a, bm, acc[mt]);
            }
        }
        int col = cb + (lane & 15);
        const float* bias = (mtx == 0) ? bq : (mtx == 1) ? bk : bv;
        float bsc = bias[col];
        if (mtx == 2) {
            bf16* dst = vo + ((size_t)b * CC + col) * HWN + hwb;
            for (int mt = 0; mt < 8; ++mt)
                for (int r = 0; r < 8; ++r) {
                    int row = mt * 16 + r + ((lane >> 4) << 3);
                    dst[row] = (bf16)(acc[mt][r] + bsc);
                }
        } else {
            bf16* dst = ((mtx == 0) ? qo : ko) + ((size_t)b * HWN + hwb) * CC + col;
            float sc = (mtx == 0) ? scale : 1.0f;
            for (int mt = 0; mt < 8; ++mt)
                for (int r = 0; r < 8; ++r) {
                    int row = mt * 16 + r + ((lane >> 4) << 3);
                    dst[(size_t)row * CC] = (bf16)((acc[mt][r] + bsc) * sc);
                }
        }
    }
}

// ---------------- Flash attention ----------------
// 512 threads (16 waves). Block: 64 queries; 256-key tiles.
// Score: wave w -> keys [w*16, w*16+16), full K=512 reduction, plain ds stores.
// Softmax: lane owns (row, 32 keys); scores exponentiated IN PLACE in sS (f32);
// PV A-fragments built by f32->bf16 conversion from sS. Wave w owns 32 channels.
__global__ void __launch_bounds__(512)
attn_kernel(const bf16* __restrict__ qm, const bf16* __restrict__ km,
            const bf16* __restrict__ vm, bf16* __restrict__ ao) {
    int b  = blockIdx.x >> 6;          // HWN/64 = 64 query tiles per batch
    int qb = (blockIdx.x & 63) * 64;
    int wave = threadIdx.x >> 5, lane = threadIdx.x & 31;
    int ch0 = wave * 32;

    __shared__ __align__(16) bf16  sQ[64 * SQLD];   // ~65KB
    __shared__ __align__(16) float sS[64 * SSLD];   // ~65KB
    __shared__ float sScale[64];

    {   // async-stage Q tile (64 x 512 bf16)
        const bf16* gq = qm + ((size_t)b * HWN + qb) * CC;
        for (int i = threadIdx.x; i < 64 * (CC / 8); i += blockDim.x) {
            int row = i >> 6, cw = i & 63;
            async_cp16(sQ + row * SQLD + cw * 8, gq + (size_t)row * CC + cw * 8);
        }
        async_wait0();
    }
    __syncthreads();

    f32x8 o[4][2];
    for (int rb = 0; rb < 4; ++rb)
        for (int cb = 0; cb < 2; ++cb) o[rb][cb] = (f32x8){};

    int srow = wave * 4 + (lane & 3);  // softmax row owned by this lane (0..63)
    int kcid = lane >> 2;              // 32-key chunk (0..7) within 256-key tile
    float m_i = -1e30f, l_i = 0.f;

    for (int nb = 0; nb < HWN; nb += 256) {
        asm volatile("" ::: "memory");  // keep sQ ds loads in-loop (no hoist/spill)

        // ---- scores: S[64 x 256] ----
        const bf16* kbase = km + ((size_t)b * HWN + nb + wave * 16) * CC;
        f32x8 s[4];
        for (int rb = 0; rb < 4; ++rb) s[rb] = (f32x8){};
        for (int kc = 0; kc < 16; ++kc) {
            bf16x16 bm = load_b_nk(kbase + kc * 32, CC, lane);
            for (int rb = 0; rb < 4; ++rb) {
                bf16x16 a = load_a_rm(sQ + rb * 16 * SQLD + kc * 32, SQLD, lane);
                s[rb] = wmma_bf16(a, bm, s[rb]);
            }
        }
        int colk = wave * 16 + (lane & 15);
        for (int rb = 0; rb < 4; ++rb)
            for (int r = 0; r < 8; ++r) {
                int row = rb * 16 + r + ((lane >> 4) << 3);
                sS[row * SSLD + colk] = s[rb][r];
            }
        if (nb + 256 < HWN) {
            __builtin_prefetch(km + ((size_t)b * HWN + nb + 256 + wave * 16 + (lane & 15)) * CC, 0, 1);
            __builtin_prefetch(vm + ((size_t)b * CC + ch0 + (lane & 15)) * HWN + nb + 256, 0, 1);
        }
        __syncthreads();

        // ---- online softmax, in place: pass 1 max, pass 2 exp+sum ----
        float* rowp = &sS[srow * SSLD + kcid * 32];
        float lmax = -1e30f;
        for (int j = 0; j < 8; ++j) {
            float4 v = ((const float4*)rowp)[j];
            lmax = fmaxf(lmax, fmaxf(fmaxf(v.x, v.y), fmaxf(v.z, v.w)));
        }
        lmax = fmaxf(lmax, __shfl_xor(lmax, 4, 32));
        lmax = fmaxf(lmax, __shfl_xor(lmax, 8, 32));
        lmax = fmaxf(lmax, __shfl_xor(lmax, 16, 32));
        float m_new = fmaxf(m_i, lmax);
        float alpha = __expf(m_i - m_new);
        float rsum = 0.f;
        for (int j = 0; j < 8; ++j) {
            float4 v = ((const float4*)rowp)[j];
            v.x = __expf(v.x - m_new); v.y = __expf(v.y - m_new);
            v.z = __expf(v.z - m_new); v.w = __expf(v.w - m_new);
            rsum += v.x + v.y + v.z + v.w;
            ((float4*)rowp)[j] = v;
        }
        rsum += __shfl_xor(rsum, 4, 32);
        rsum += __shfl_xor(rsum, 8, 32);
        rsum += __shfl_xor(rsum, 16, 32);
        l_i = l_i * alpha + rsum;
        m_i = m_new;
        if (kcid == 0) sScale[srow] = alpha;
        __syncthreads();

        // ---- rescale O by per-row alpha ----
        for (int rb = 0; rb < 4; ++rb) {
            float a8[8];
            for (int r = 0; r < 8; ++r)
                a8[r] = sScale[rb * 16 + r + ((lane >> 4) << 3)];
            for (int cb = 0; cb < 2; ++cb)
                for (int r = 0; r < 8; ++r) o[rb][cb][r] *= a8[r];
        }
        // ---- PV: O[q][ch] += P[q][k] * V[ch][k], wave's 32-channel slice ----
        for (int kc = 0; kc < 8; ++kc) {
            bf16x16 pa[4];
            for (int rb = 0; rb < 4; ++rb)
                pa[rb] = load_a_f32(sS + rb * 16 * SSLD + kc * 32, SSLD, lane);
            for (int cb = 0; cb < 2; ++cb) {
                bf16x16 bv_ = load_b_nk(
                    vm + ((size_t)b * CC + ch0 + cb * 16) * HWN + nb + kc * 32,
                    HWN, lane);
                for (int rb = 0; rb < 4; ++rb)
                    o[rb][cb] = wmma_bf16(pa[rb], bv_, o[rb][cb]);
            }
        }
        __syncthreads();   // sS reused by next iteration's score stores
    }

    if (kcid == 0) sScale[srow] = l_i;   // final denominators (post loop barrier)
    __syncthreads();
    for (int rb = 0; rb < 4; ++rb) {
        float inv[8];
        for (int r = 0; r < 8; ++r)
            inv[r] = 1.0f / sScale[rb * 16 + r + ((lane >> 4) << 3)];
        for (int cb = 0; cb < 2; ++cb) {
            int col = ch0 + cb * 16 + (lane & 15);
            bf16* dst = ao + ((size_t)b * HWN + qb + rb * 16) * CC + col;
            for (int r = 0; r < 8; ++r) {
                int row = r + ((lane >> 4) << 3);
                dst[(size_t)row * CC] = (bf16)(o[rb][cb][r] * inv[r]);
            }
        }
    }
}

// ---------------- Output projection + residual (M = 128 rows / block) ----------------
__global__ void proj_kernel(const bf16* __restrict__ ao, const bf16* __restrict__ wbp,
                            const float* __restrict__ bp, const float* __restrict__ feat,
                            float* __restrict__ out) {
    int blk = blockIdx.x;
    int b   = blk >> 5;
    int hwb = (blk & 31) * 128;
    __shared__ __align__(16) bf16 sA[128 * SALD];
    {
        const bf16* g = ao + ((size_t)b * HWN + hwb) * CC;
        for (int i = threadIdx.x; i < 128 * (CC / 8); i += blockDim.x) {
            int row = i >> 6, cw = i & 63;
            async_cp16(sA + row * SALD + cw * 8, g + (size_t)row * CC + cw * 8);
        }
        async_wait0();
    }
    __syncthreads();
    int wave = threadIdx.x >> 5, lane = threadIdx.x & 31;
    for (int tt = 0; tt < 4; ++tt) {          // 32 chains over 8 waves
        asm volatile("" ::: "memory");
        int cb = (wave + tt * 8) * 16;
        const bf16* W = wbp + (size_t)cb * CC;
        f32x8 acc[8];
        for (int mt = 0; mt < 8; ++mt) acc[mt] = (f32x8){};
        for (int kb = 0; kb < CC; kb += 32) {
            bf16x16 bm = load_b_nk(W + kb, CC, lane);
            for (int mt = 0; mt < 8; ++mt) {
                bf16x16 a = load_a_rm(sA + mt * 16 * SALD + kb, SALD, lane);
                acc[mt] = wmma_bf16(a, bm, acc[mt]);
            }
        }
        int col = cb + (lane & 15);
        float bsc = bp[col];
        const float* fsrc = feat + ((size_t)b * CC + col) * HWN + hwb;
        float*       dst  = out  + ((size_t)b * CC + col) * HWN + hwb;
        for (int mt = 0; mt < 8; ++mt)
            for (int r = 0; r < 8; ++r) {
                int row = mt * 16 + r + ((lane >> 4) << 3);
                dst[row] = fsrc[row] + acc[mt][r] + bsc;
            }
    }
}

extern "C" void kernel_launch(void* const* d_in, const int* in_sizes, int n_in,
                              void* d_out, int out_size, void* d_ws, size_t ws_size,
                              hipStream_t stream) {
    (void)in_sizes; (void)n_in; (void)out_size; (void)ws_size;
    const float* feature = (const float*)d_in[0];
    const float* gamma   = (const float*)d_in[1];
    const float* beta    = (const float*)d_in[2];
    const float* wq      = (const float*)d_in[3];
    const float* bq      = (const float*)d_in[4];
    const float* wk      = (const float*)d_in[5];
    const float* bk      = (const float*)d_in[6];
    const float* wv      = (const float*)d_in[7];
    const float* bv      = (const float*)d_in[8];
    const float* wp      = (const float*)d_in[9];
    const float* bp      = (const float*)d_in[10];
    float* out = (float*)d_out;

    const size_t NBHWC = (size_t)BB * HWN * CC;   // 8,388,608 elements
    bf16* xn = (bf16*)d_ws;
    bf16* qo = xn + NBHWC;
    bf16* ko = qo + NBHWC;
    bf16* vo = ko + NBHWC;
    bf16* ao = vo + NBHWC;
    bf16* wb = ao + NBHWC;                        // 4 * CC*CC bf16
    float* stats = (float*)(wb + (size_t)4 * CC * CC);

    wcvt_kernel<<<(4 * CC * CC + 255) / 256, 256, 0, stream>>>(wq, wk, wv, wp, wb);
    gn_stats_kernel<<<BB * GR, 256, 0, stream>>>(feature, stats);
    gn_apply_kernel<<<BB * CC, 256, 0, stream>>>(feature, gamma, beta, stats, xn);
    qkv_kernel<<<BB * (HWN / 128), 256, 0, stream>>>(xn, wb, bq, bk, bv, qo, ko, vo);
    attn_kernel<<<BB * (HWN / 64), 512, 0, stream>>>(qo, ko, vo, ao);
    proj_kernel<<<BB * (HWN / 128), 256, 0, stream>>>(ao, wb + (size_t)3 * CC * CC, bp,
                                                      feature, out);
}
